// QTemporal_6107443495210
// MI455X (gfx1250) — matrix operand order repaired and checked by reading, not compiled, not run on hardware
//
#include <hip/hip_runtime.h>

// ---------------------------------------------------------------------------
// Problem constants (match reference)
// ---------------------------------------------------------------------------
#define B_DIM   64
#define T_DIM   2048
#define F_DIM   64
#define K_DIM   512
#define D_DIM   512
#define T_TILE  64
#define N_TILES (T_DIM / T_TILE)     // 32
#define Y2_PITCH 513                 // f32 pitch, odd mod 64 -> conflict-free
#define LN_EPS  1e-5f

typedef __attribute__((ext_vector_type(16))) __bf16          v16bf;
typedef __attribute__((ext_vector_type(8)))  float           v8f;
typedef __attribute__((ext_vector_type(8)))  unsigned short  su8;
typedef __attribute__((ext_vector_type(16))) unsigned short  su16;

__device__ __forceinline__ unsigned short f2bf(float f) {
    unsigned u = __builtin_bit_cast(unsigned, f);
    u += 0x7FFFu + ((u >> 16) & 1u);        // round-to-nearest-even
    return (unsigned short)(u >> 16);
}

// A-matrix 16x32 bf16 fragment: per lane two contiguous 8-halfword chunks
// (chunk0 at k0, chunk1 at k0+8 halfwords). 2x ds_load_b128 when p is LDS.
__device__ __forceinline__ v16bf frag_from2(const unsigned short* p) {
    su8 lo = *(const su8*)p;
    su8 hi = *(const su8*)(p + 8);
    su16 t = __builtin_shufflevector(lo, hi,
        0,1,2,3,4,5,6,7,8,9,10,11,12,13,14,15);
    return __builtin_bit_cast(v16bf, t);
}

// B-matrix 32x16 bf16 fragment: per lane 16 contiguous K halfwords (32 bytes).
__device__ __forceinline__ v16bf frag_contig(const unsigned short* p) {
    su16 t = *(const su16*)p;
    return __builtin_bit_cast(v16bf, t);
}

__device__ __forceinline__ v8f wmma_bf16(v16bf a, v16bf b, v8f c) {
    return __builtin_amdgcn_wmma_f32_16x16x32_bf16(
        false, a, false, b, (short)0, c, false, false);
}

// Async global->LDS copy of 16 bytes (gfx1250 GLOBAL_LOAD_ASYNC_TO_LDS_B128,
// tracked with ASYNCcnt). ldsoff is the workgroup-relative LDS byte address
// (low 32 bits of the generic LDS pointer).
__device__ __forceinline__ void async_g2l_b128(unsigned ldsoff,
                                               unsigned long long gaddr) {
    asm volatile("global_load_async_to_lds_b128 %0, %1, off"
                 :: "v"(ldsoff), "v"(gaddr) : "memory");
}
__device__ __forceinline__ void wait_asynccnt0() {
    asm volatile("s_wait_asynccnt 0x0" ::: "memory");
}

// ---------------------------------------------------------------------------
// Prep: convert P_w [512,64] and W_w [512,512] f32 -> bf16 in workspace
// ---------------------------------------------------------------------------
__global__ void qt_prep_bf16(const float* __restrict__ P,
                             const float* __restrict__ W,
                             unsigned short* __restrict__ Pbf,
                             unsigned short* __restrict__ Wbf) {
    int i = blockIdx.x * blockDim.x + threadIdx.x;
    if (i < K_DIM * F_DIM) Pbf[i] = f2bf(P[i]);
    if (i < D_DIM * K_DIM) Wbf[i] = f2bf(W[i]);
}

// ---------------------------------------------------------------------------
// Fused: x-tile -> GEMM1 -> GEMM2 -> bias+LN -> EMA-weighted tile partial
// grid = (32 tiles, 64 batches), block = 256 (8 waves)
// ---------------------------------------------------------------------------
__global__ __launch_bounds__(256, 1)
void qt_fused(const float* __restrict__ x,
              const float* __restrict__ th1, const float* __restrict__ ph1,
              const float* __restrict__ th2, const float* __restrict__ ph2,
              const float* __restrict__ w1p, const float* __restrict__ w2p,
              const float* __restrict__ bgp,
              const unsigned short* __restrict__ Pbf,
              const unsigned short* __restrict__ Wbf,
              const float* __restrict__ Wb,   const float* __restrict__ bb,
              const float* __restrict__ gam,  const float* __restrict__ bet,
              float* __restrict__ partial) {
    extern __shared__ char smem[];
    unsigned short* xs = (unsigned short*)smem;                       // 64x64 bf16 (8 KB)
    unsigned short* y1 = (unsigned short*)(smem + 8192);              // 64x512 bf16 (64 KB)
    float*          y2 = (float*)(smem + 8192 + 65536);               // 64x513 f32 (~128 KB)

    const int tid  = threadIdx.x;
    const int tile = blockIdx.x;
    const int bidx = blockIdx.y;

    // ---- Phase 1: async-stage raw f32 x tile into LDS (y2 region is dead
    //      until phase 3, reuse it as the staging buffer), then convert bf16.
    {
        float* stage = y2;                                   // 16 KB staging
        const char* xg = (const char*)(x +
            ((size_t)bidx * T_DIM + (size_t)tile * T_TILE) * F_DIM);
        unsigned stage_off = (unsigned)(unsigned long long)(void*)stage;
        #pragma unroll
        for (int i = 0; i < 4; ++i) {
            unsigned boff = (unsigned)(tid + i * 256) * 16u;  // 1024 x 16B
            async_g2l_b128(stage_off + boff,
                           (unsigned long long)(xg + boff));
        }
        wait_asynccnt0();
        __syncthreads();

        unsigned* xsu = (unsigned*)xs;
        const float4* sf = (const float4*)stage;
        #pragma unroll
        for (int i = 0; i < 4; ++i) {
            int j = tid + i * 256;
            float4 v = sf[j];
            xsu[2 * j + 0] = (unsigned)f2bf(v.x) | ((unsigned)f2bf(v.y) << 16);
            xsu[2 * j + 1] = (unsigned)f2bf(v.z) | ((unsigned)f2bf(v.w) << 16);
        }
    }
    __syncthreads();

    const int wv    = tid >> 5;
    const int lane  = tid & 31;
    const int mblk  = wv & 3;                  // 4 row blocks of 16
    const int nh    = wv >> 2;                 // 0/1: halves of the 32 N-blocks
    const int lrow  = lane & 15;
    const int ksel  = (lane < 16) ? 0 : 16;    // A-fragment K offset
    const int bksel = (lane < 16) ? 0 : 16;    // B-fragment K offset
    const int lcol  = lane & 15;
    const int rowo  = mblk * 16 + ((lane < 16) ? 0 : 8);

    // ---- Phase 2: GEMM1  y1[64,512] = x_tile @ P_w^T ----
    {
        const unsigned short* arow = xs + (mblk * 16 + lrow) * F_DIM;
        v16bf xa0 = frag_from2(arow + 0  + ksel);
        v16bf xa1 = frag_from2(arow + 32 + ksel);
        for (int nbi = 0; nbi < 16; ++nbi) {
            const int n    = nh * 16 + nbi;          // N-block over K_DIM
            const int ncol = n * 16 + lcol;
            const unsigned short* prow = Pbf + (size_t)ncol * F_DIM;
            v8f acc = {};
            acc = wmma_bf16(xa0, frag_contig(prow + 0  + bksel), acc);
            acc = wmma_bf16(xa1, frag_contig(prow + 32 + bksel), acc);
            #pragma unroll
            for (int j = 0; j < 8; ++j)
                y1[(size_t)(rowo + j) * K_DIM + ncol] = f2bf(acc[j]);
        }
    }
    __syncthreads();

    // ---- Phase 3: GEMM2  y2[64,512] = y1 @ W_w^T ----
    {
        v16bf A[16];
        const unsigned short* arow = y1 + (size_t)(mblk * 16 + lrow) * K_DIM;
        #pragma unroll
        for (int kb = 0; kb < 16; ++kb)
            A[kb] = frag_from2(arow + kb * 32 + ksel);

        for (int nbi = 0; nbi < 16; ++nbi) {
            const int dcol = (nh * 16 + nbi) * 16 + lcol;
            const unsigned short* wrow = Wbf + (size_t)dcol * K_DIM;
            v8f acc = {};
            #pragma unroll
            for (int kb = 0; kb < 16; ++kb)
                acc = wmma_bf16(A[kb], frag_contig(wrow + kb * 32 + bksel), acc);
            #pragma unroll
            for (int j = 0; j < 8; ++j)
                y2[(size_t)(rowo + j) * Y2_PITCH + dcol] = acc[j];
        }
    }
    __syncthreads();

    // ---- Phase 4a: bias + LayerNorm + EMA weight ----
    // One wave per 8 rows; lanes split the 512 features, butterfly-reduce.
    {
        // scalar gate (alpha*cal_scalar cancels through LayerNorm exactly)
        float z1 = __cosf(th1[0]) * __cosf(ph1[0]);
        float z2 = __cosf(th2[0]) * __cosf(ph2[0]);
        float s  = w1p[0] * z1 + w2p[0] * z2 + bgp[0];
        float g  = 1.0f / (1.0f + __expf(-s));
        g = fminf(0.95f, fmaxf(0.05f, g));
        const float omg = 1.0f - g;

        const int r0 = wv * 8;
        for (int rr = 0; rr < 8; ++rr) {
            const int r = r0 + rr;
            float* row = y2 + (size_t)r * Y2_PITCH;
            float sum = 0.f, sumsq = 0.f;
            #pragma unroll
            for (int i = 0; i < 16; ++i) {
                int d = lane + i * 32;
                float v = row[d] + Wb[d] + bb[d];
                row[d] = v;
                sum   += v;
                sumsq += v * v;
            }
            #pragma unroll
            for (int sft = 16; sft > 0; sft >>= 1) {
                sum   += __shfl_xor(sum,   sft, 32);
                sumsq += __shfl_xor(sumsq, sft, 32);
            }
            const float inv  = 1.0f / (float)D_DIM;
            float mu   = sum * inv;
            float var  = sumsq * inv - mu * mu;
            float rstd = rsqrtf(var + LN_EPS);

            // h = sum_t g*(1-g)^(T-1-t) * LN(base_t): fold weight into row
            int   tg = tile * T_TILE + r;
            float wr = g * __powf(omg, (float)(T_DIM - 1 - tg));
            #pragma unroll
            for (int i = 0; i < 16; ++i) {
                int d = lane + i * 32;
                row[d] = wr * ((row[d] - mu) * rstd * gam[d] + bet[d]);
            }
        }
    }
    __syncthreads();

    // ---- Phase 4b: column sums -> per-(b,tile) partial (deterministic) ----
    #pragma unroll
    for (int c = 0; c < 2; ++c) {
        int d = tid + c * 256;
        float s = 0.f;
        for (int r = 0; r < T_TILE; ++r)
            s += y2[(size_t)r * Y2_PITCH + d];
        partial[((size_t)bidx * N_TILES + tile) * D_DIM + d] = s;
    }
}

// ---------------------------------------------------------------------------
// Reduce: out[b,d] = sum over 32 tile partials
// ---------------------------------------------------------------------------
__global__ void qt_reduce(const float* __restrict__ partial,
                          float* __restrict__ out) {
    int i = blockIdx.x * blockDim.x + threadIdx.x;
    if (i >= B_DIM * D_DIM) return;
    int b = i >> 9;            // /512
    int d = i & (D_DIM - 1);
    const float* p = partial + ((size_t)b * N_TILES) * D_DIM + d;
    float s = 0.f;
    #pragma unroll
    for (int t = 0; t < N_TILES; ++t) s += p[(size_t)t * D_DIM];
    out[i] = s;
}

// ---------------------------------------------------------------------------
extern "C" void kernel_launch(void* const* d_in, const int* in_sizes, int n_in,
                              void* d_out, int out_size, void* d_ws, size_t ws_size,
                              hipStream_t stream) {
    const float* x    = (const float*)d_in[0];
    const float* th1  = (const float*)d_in[1];
    const float* ph1  = (const float*)d_in[2];
    const float* th2  = (const float*)d_in[3];
    const float* ph2  = (const float*)d_in[4];
    const float* w1p  = (const float*)d_in[5];
    const float* w2p  = (const float*)d_in[6];
    const float* bgp  = (const float*)d_in[7];
    const float* P_w  = (const float*)d_in[8];
    const float* W_w  = (const float*)d_in[9];
    const float* W_b  = (const float*)d_in[10];
    const float* bv   = (const float*)d_in[11];
    const float* gam  = (const float*)d_in[12];
    const float* bet  = (const float*)d_in[13];
    // d_in[14] = alpha: mathematically cancels through LayerNorm (constant over D)

    unsigned short* Pbf = (unsigned short*)d_ws;                              // 64 KB
    unsigned short* Wbf = (unsigned short*)((char*)d_ws + 65536);             // 512 KB
    float* partial      = (float*)((char*)d_ws + 65536 + 524288);             // 4 MB

    qt_prep_bf16<<<dim3((D_DIM * K_DIM + 255) / 256), dim3(256), 0, stream>>>(
        P_w, W_w, Pbf, Wbf);

    const size_t lds_bytes = 8192 + 65536 + (size_t)T_TILE * Y2_PITCH * 4;    // 205056
    hipFuncSetAttribute((const void*)qt_fused,
                        hipFuncAttributeMaxDynamicSharedMemorySize,
                        (int)lds_bytes);
    qt_fused<<<dim3(N_TILES, B_DIM), dim3(256), lds_bytes, stream>>>(
        x, th1, ph1, th2, ph2, w1p, w2p, bgp,
        Pbf, Wbf, W_b, bv, gam, bet, partial);

    qt_reduce<<<dim3((B_DIM * D_DIM + 255) / 256), dim3(256), 0, stream>>>(
        partial, (float*)d_out);
}
